// c_5_75445395521981
// MI455X (gfx1250) — matrix operand order repaired and checked by reading, not compiled, not run on hardware
//
#include <hip/hip_runtime.h>

typedef _Float16 half_t;
typedef __attribute__((ext_vector_type(16))) _Float16 v16h;
typedef __attribute__((ext_vector_type(8)))  float    v8f;

#define NB 32           // batch
#define NIMG 1280       // 2 modalities * 32 * 5 * 4

// ---------------------------------------------------------------------------
// Stage 0: convert MLP weights to f16 stored in WMMA B-fragment order.
// For each 32(K) x 16(N) tile, 512 halves are stored contiguously as
// 32 lanes x 16 halves:  lane<16  -> B[k0+0..15 ][n0+lane]
//                        lane>=16 -> B[k0+16..31][n0+lane-16]
// so loadB is a single aligned 32-byte vector load per lane.
//   o1w [360,180] -> w1: 12x12 tiles (73728 halves)
//   o2w [180, 60] -> w2:  6x 4 tiles (12288 halves)
//   o3w [ 60,  6] -> w3:  2x 1 tiles ( 1024 halves)
// ---------------------------------------------------------------------------
__global__ void cvt_weights(const float* __restrict__ o1w,
                            const float* __restrict__ o2w,
                            const float* __restrict__ o3w,
                            half_t* __restrict__ w1,
                            half_t* __restrict__ w2,
                            half_t* __restrict__ w3) {
    int idx = blockIdx.x * blockDim.x + threadIdx.x;
    if (idx < 73728) {
        int frag = idx >> 9, r = idx & 511;
        int lane = r >> 4, i = r & 15;
        int kc = frag / 12, nt = frag % 12;
        int k = kc * 32 + ((lane & 16) ? 16 : 0) + i;
        int n = nt * 16 + (lane & 15);
        w1[idx] = (half_t)((k < 360 && n < 180) ? o1w[k * 180 + n] : 0.f);
    } else if (idx < 73728 + 12288) {
        int j = idx - 73728;
        int frag = j >> 9, r = j & 511;
        int lane = r >> 4, i = r & 15;
        int kc = frag >> 2, nt = frag & 3;
        int k = kc * 32 + ((lane & 16) ? 16 : 0) + i;
        int n = nt * 16 + (lane & 15);
        w2[j] = (half_t)((k < 180 && n < 60) ? o2w[k * 60 + n] : 0.f);
    } else if (idx < 73728 + 12288 + 1024) {
        int j = idx - (73728 + 12288);
        int frag = j >> 9, r = j & 511;
        int lane = r >> 4, i = r & 15;
        int k = frag * 32 + ((lane & 16) ? 16 : 0) + i;
        int n = lane & 15;
        w3[j] = (half_t)((k < 60 && n < 6) ? o3w[k * 6 + n] : 0.f);
    }
}

// ---------------------------------------------------------------------------
// Stage 1: fused conv1(3->3, 3x3, s2) + bias + maxpool(3,3) + relu
// grid = (37 pool rows, 1280 images), block = 128
// Streams each input byte from HBM exactly once (memory-bound stage:
// ~770 MB @ 23.3 TB/s ~= 33 us floor).
// ---------------------------------------------------------------------------
__global__ __launch_bounds__(128)
void conv1_pool(const float* __restrict__ nodes, const float* __restrict__ depths,
                const float* __restrict__ c1w, const float* __restrict__ c1b,
                const float* __restrict__ dc1w, const float* __restrict__ dc1b,
                float* __restrict__ s1) {
    const int r   = blockIdx.x;   // pool row 0..36
    const int img = blockIdx.y;   // 0..1279
    const bool rgb = img < 640;
    const float* src = rgb ? nodes  + (size_t)img * (3 * 224 * 224)
                           : depths + (size_t)(img - 640) * (3 * 224 * 224);
    const float* w = rgb ? c1w : dc1w;
    const float* b = rgb ? c1b : dc1b;

    __shared__ float tile[3 * 7 * 224];  // [ic][row][col], input rows 6r..6r+6
    __shared__ float wS[81];
    __shared__ float bS[3];

    const int tid = threadIdx.x;
    if (tid < 81) wS[tid] = w[tid];
    if (tid < 3)  bS[tid] = b[tid];

    const int row0 = 6 * r;
    // 21 segments (ic,row) of 224 floats = 56 float4 each -> 1176 float4 loads
    for (int idx = tid; idx < 21 * 56; idx += 128) {
        int seg = idx / 56, off = (idx % 56) * 4;
        int c = seg / 7, rr = seg % 7;
        float4 v = *(const float4*)(src + (size_t)c * (224 * 224) + (row0 + rr) * 224 + off);
        *(float4*)(tile + seg * 224 + off) = v;
    }
    __syncthreads();

    if (tid < 111) {
        int oc = tid / 37, pc = tid % 37;
        float best = -1e30f;
        #pragma unroll
        for (int i = 0; i < 3; ++i) {
            #pragma unroll
            for (int j = 0; j < 3; ++j) {
                float acc = bS[oc];
                int ry = 2 * i;               // LDS row of conv row 3r+i
                int cx = 2 * (3 * pc + j);    // input col base
                #pragma unroll
                for (int ic = 0; ic < 3; ++ic)
                    #pragma unroll
                    for (int ky = 0; ky < 3; ++ky)
                        #pragma unroll
                        for (int kx = 0; kx < 3; ++kx)
                            acc += wS[(oc * 3 + ic) * 9 + ky * 3 + kx] *
                                   tile[(ic * 7 + ry + ky) * 224 + cx + kx];
                best = fmaxf(best, acc);
            }
        }
        s1[((size_t)img * 3 + oc) * 1369 + r * 37 + pc] = fmaxf(best, 0.f);
    }
}

// ---------------------------------------------------------------------------
// Stage 2: conv2(3->1, 3x3, s2) + bias + maxpool(3,3) + relu + linear(36->6)
// grid = 1280 images, block = 128
// ---------------------------------------------------------------------------
__global__ __launch_bounds__(128)
void conv2_pool_lin(const float* __restrict__ s1,
                    const float* __restrict__ c2w, const float* __restrict__ c2b,
                    const float* __restrict__ lw,  const float* __restrict__ lb,
                    const float* __restrict__ dc2w, const float* __restrict__ dc2b,
                    const float* __restrict__ dlw,  const float* __restrict__ dlb,
                    float* __restrict__ nfdf) {
    const int img = blockIdx.x;
    const bool rgb = img < 640;
    const float* w  = rgb ? c2w : dc2w;
    const float* bb = rgb ? c2b : dc2b;
    const float* L  = rgb ? lw  : dlw;
    const float* Lb = rgb ? lb  : dlb;

    __shared__ float tile[3 * 37 * 37];   // 4107
    __shared__ float h[36];
    __shared__ float wS[27];
    __shared__ float bS;
    __shared__ float LwS[216];
    __shared__ float LbS[6];

    int tid = threadIdx.x;
    if (tid < 27) wS[tid] = w[tid];
    if (tid == 27) bS = bb[0];
    for (int i = tid; i < 216; i += 128) LwS[i] = L[i];
    if (tid < 6) LbS[tid] = Lb[tid];
    for (int i = tid; i < 4107; i += 128) tile[i] = s1[(size_t)img * 4107 + i];
    __syncthreads();

    if (tid < 36) {
        int pr = tid / 6, pc = tid % 6;
        float best = -1e30f;
        #pragma unroll
        for (int i = 0; i < 3; ++i) {
            #pragma unroll
            for (int j = 0; j < 3; ++j) {
                int cr = 3 * pr + i, cc = 3 * pc + j;
                float acc = bS;
                #pragma unroll
                for (int ic = 0; ic < 3; ++ic)
                    #pragma unroll
                    for (int ky = 0; ky < 3; ++ky)
                        #pragma unroll
                        for (int kx = 0; kx < 3; ++kx)
                            acc += wS[ic * 9 + ky * 3 + kx] *
                                   tile[ic * 1369 + (2 * cr + ky) * 37 + (2 * cc + kx)];
                best = fmaxf(best, acc);
            }
        }
        h[tid] = fmaxf(best, 0.f);
    }
    __syncthreads();
    if (tid < 6) {
        float acc = LbS[tid];
        #pragma unroll
        for (int j = 0; j < 36; ++j) acc += h[j] * LwS[j * 6 + tid];
        nfdf[(size_t)img * 6 + tid] = acc;
    }
}

// ---------------------------------------------------------------------------
// Stage 3: message passing + feat assembly + 3-layer MLP via WMMA (f16 in, f32 acc)
// grid = 1 block of 256 threads (8 waves)
// ---------------------------------------------------------------------------
// A fragment (16x32 f16) from LDS: lane<16 holds row m0+lane, K = k0+{0..15};
//                                  lane>=16 holds row m0+lane-16, K = k0+{16..31}
__device__ inline v16h loadA(const half_t* A, int lda, int m0, int k0, int lane) {
    v16h a;
    int row = m0 + (lane & 15);
    int kb  = k0 + ((lane & 16) ? 16 : 0);
    #pragma unroll
    for (int i = 0; i < 8; ++i) a[i]     = A[row * lda + kb + i];
    #pragma unroll
    for (int i = 0; i < 8; ++i) a[8 + i] = A[row * lda + kb + 8 + i];
    return a;
}
// B fragment: pre-swizzled in global memory -> one 32-byte vector load per lane
__device__ inline v16h loadBfrag(const half_t* __restrict__ frags, int fragIdx, int lane) {
    return ((const v16h*)frags)[fragIdx * 32 + lane];
}

__global__ __launch_bounds__(256)
void head(const float* __restrict__ pos, const float* __restrict__ att,
          const float* __restrict__ fmw, const float* __restrict__ fmb,
          const float* __restrict__ lmw, const float* __restrict__ lmb,
          const float* __restrict__ o1b, const float* __restrict__ o2b,
          const float* __restrict__ o3b,
          const float* __restrict__ nf, const float* __restrict__ df,
          const half_t* __restrict__ w1, const half_t* __restrict__ w2,
          const half_t* __restrict__ w3,
          float* __restrict__ out) {
    __shared__ half_t featS[32 * 384];   // feat, K padded 360->384
    __shared__ half_t h1S[32 * 192];     // layer1 out, N padded 180->192
    __shared__ half_t h2S[32 * 64];      // layer2 out, N padded 60->64
    __shared__ float  puS[32 * 120];     // pos_up [b][(f*4+n)*6+d]

    const int tid  = threadIdx.x;
    const int wave = tid >> 5;
    const int lane = tid & 31;

    for (int i = tid; i < 32 * 192; i += 256) h1S[i] = (half_t)0.f;
    for (int i = tid; i < 32 * 64;  i += 256) h2S[i] = (half_t)0.f;

    // ---- message passing: pos_up = einsum(att, pos@fmw+fmb) + shift(pos@lmw+lmb)
    for (int idx = tid; idx < NB * 120; idx += 256) {
        int b = idx / 120, rem = idx % 120;
        int q = rem / 6, d = rem % 6;   // q = f*4+n
        int f = q >> 2, n = q & 3;
        const float* posb = pos + (size_t)(b * 5 + f) * 24;
        const float* attb = att + (size_t)(b * 5 + f) * 16;
        float acc = 0.f;
        #pragma unroll
        for (int m = 0; m < 4; ++m) {
            float pm = fmb[d];
            #pragma unroll
            for (int k = 0; k < 6; ++k) pm += posb[m * 6 + k] * fmw[k * 6 + d];
            acc += attb[m * 4 + n] * pm;
        }
        if (f >= 1) {
            const float* posp = pos + (size_t)(b * 5 + f - 1) * 24;
            float pv = lmb[d];
            #pragma unroll
            for (int k = 0; k < 6; ++k) pv += posp[n * 6 + k] * lmw[k * 6 + d];
            acc += pv;
        }
        puS[b * 120 + rem] = acc;
    }
    __syncthreads();

    // ---- feat assembly: [nf | df | pos_up] per node -> [32,384] f16 (pad 0)
    for (int idx = tid; idx < 32 * 384; idx += 256) {
        int b = idx / 384, j = idx % 384;
        float v = 0.f;
        if (j < 360) {
            int q = j / 18, c = j % 18;     // q = f*4+n
            int gi = b * 20 + q;            // (b*5+f)*4+n
            if (c < 6)       v = nf[gi * 6 + c];
            else if (c < 12) v = df[gi * 6 + (c - 6)];
            else             v = puS[b * 120 + q * 6 + (c - 12)];
        }
        featS[idx] = (half_t)v;
    }
    __syncthreads();

    // ---- GEMM1: [32,384] @ [384,192] -> relu -> h1S   (24 tiles / 8 waves)
    for (int t = wave; t < 24; t += 8) {
        int mt = t / 12, nt = t % 12;
        int m0 = mt * 16, n0 = nt * 16;
        v8f acc = {};
        #pragma unroll 4
        for (int kc = 0; kc < 12; ++kc) {
            v16h a = loadA(featS, 384, m0, kc * 32, lane);
            v16h b = loadBfrag(w1, kc * 12 + nt, lane);
            acc = __builtin_amdgcn_wmma_f32_16x16x32_f16(false, a, false, b,
                                                         (short)0, acc, false, false);
        }
        int col   = n0 + (lane & 15);
        int rbase = m0 + ((lane & 16) ? 8 : 0);
        if (col < 180) {
            float bias = o1b[col];
            #pragma unroll
            for (int r = 0; r < 8; ++r)
                h1S[(rbase + r) * 192 + col] = (half_t)fmaxf(acc[r] + bias, 0.f);
        }
    }
    __syncthreads();

    // ---- GEMM2: [32,192] @ [192,64] -> relu -> h2S   (8 tiles / 8 waves)
    {
        int m0 = (wave >> 2) * 16, n0 = (wave & 3) * 16;
        int nt = wave & 3;
        v8f acc = {};
        #pragma unroll
        for (int kc = 0; kc < 6; ++kc) {
            v16h a = loadA(h1S, 192, m0, kc * 32, lane);
            v16h b = loadBfrag(w2, kc * 4 + nt, lane);
            acc = __builtin_amdgcn_wmma_f32_16x16x32_f16(false, a, false, b,
                                                         (short)0, acc, false, false);
        }
        int col   = n0 + (lane & 15);
        int rbase = m0 + ((lane & 16) ? 8 : 0);
        if (col < 60) {
            float bias = o2b[col];
            #pragma unroll
            for (int r = 0; r < 8; ++r)
                h2S[(rbase + r) * 64 + col] = (half_t)fmaxf(acc[r] + bias, 0.f);
        }
    }
    __syncthreads();

    // ---- GEMM3: [32,64] @ [64,16] -> out[32,6]   (2 tiles, waves 0-1)
    if (wave < 2) {
        int m0 = wave * 16;
        v8f acc = {};
        #pragma unroll
        for (int kc = 0; kc < 2; ++kc) {
            v16h a = loadA(h2S, 64, m0, kc * 32, lane);
            v16h b = loadBfrag(w3, kc, lane);
            acc = __builtin_amdgcn_wmma_f32_16x16x32_f16(false, a, false, b,
                                                         (short)0, acc, false, false);
        }
        int col   = lane & 15;
        int rbase = m0 + ((lane & 16) ? 8 : 0);
        if (col < 6) {
            float bias = o3b[col];
            #pragma unroll
            for (int r = 0; r < 8; ++r)
                out[(rbase + r) * 6 + col] = acc[r] + bias;
        }
    }
}

// ---------------------------------------------------------------------------
extern "C" void kernel_launch(void* const* d_in, const int* in_sizes, int n_in,
                              void* d_out, int out_size, void* d_ws, size_t ws_size,
                              hipStream_t stream) {
    const float* nodes  = (const float*)d_in[0];
    const float* pos    = (const float*)d_in[1];
    const float* att    = (const float*)d_in[2];
    const float* depths = (const float*)d_in[3];
    const float* c1w  = (const float*)d_in[4];
    const float* c1b  = (const float*)d_in[5];
    const float* c2w  = (const float*)d_in[6];
    const float* c2b  = (const float*)d_in[7];
    const float* lw   = (const float*)d_in[8];
    const float* lb   = (const float*)d_in[9];
    const float* dc1w = (const float*)d_in[10];
    const float* dc1b = (const float*)d_in[11];
    const float* dc2w = (const float*)d_in[12];
    const float* dc2b = (const float*)d_in[13];
    const float* dlw  = (const float*)d_in[14];
    const float* dlb  = (const float*)d_in[15];
    const float* fmw  = (const float*)d_in[16];
    const float* fmb  = (const float*)d_in[17];
    const float* lmw  = (const float*)d_in[18];
    const float* lmb  = (const float*)d_in[19];
    const float* o1w  = (const float*)d_in[20];
    const float* o1b  = (const float*)d_in[21];
    const float* o2w  = (const float*)d_in[22];
    const float* o2b  = (const float*)d_in[23];
    const float* o3w  = (const float*)d_in[24];
    const float* o3b  = (const float*)d_in[25];

    float* ws   = (float*)d_ws;
    float* s1   = ws;                             // [1280][3][37][37] = 5,256,960 f32
    float* nfdf = ws + (size_t)NIMG * 4107;       // [1280][6]
    half_t* w1  = (half_t*)(nfdf + NIMG * 6);     // 144 B-fragments of 512 halves
    half_t* w2  = w1 + 73728;                     //  24 B-fragments
    half_t* w3  = w2 + 12288;                     //   2 B-fragments

    cvt_weights<<<340, 256, 0, stream>>>(o1w, o2w, o3w, w1, w2, w3);
    conv1_pool<<<dim3(37, NIMG), 128, 0, stream>>>(nodes, depths, c1w, c1b, dc1w, dc1b, s1);
    conv2_pool_lin<<<NIMG, 128, 0, stream>>>(s1, c2w, c2b, lw, lb, dc2w, dc2b, dlw, dlb, nfdf);
    head<<<1, 256, 0, stream>>>(pos, att, fmw, fmb, lmw, lmb, o1b, o2b, o3b,
                                nfdf, nfdf + 640 * 6, w1, w2, w3, (float*)d_out);

    (void)in_sizes; (void)n_in; (void)out_size; (void)ws_size;
}
